// EntityDetection_54202487275883
// MI455X (gfx1250) — matrix operand order repaired and checked by reading, not compiled, not run on hardware
//
#include <hip/hip_runtime.h>
#include <math.h>

typedef __bf16 bf16;
typedef __attribute__((ext_vector_type(16))) __bf16 v16bf;
typedef __attribute__((ext_vector_type(8)))  __bf16 v8bf;
typedef __attribute__((ext_vector_type(4)))  __bf16 v4bf;
typedef __attribute__((ext_vector_type(8)))  float  v8f;
typedef __attribute__((ext_vector_type(4)))  float  v4f;

#define Bb   128
#define Ll   512
#define Dd   512
#define Ee   64
#define Hh   512
#define Vv   17
#define KT   (Dd + Ee)     // 576
#define PADi 0

#define AST  600           // sA row stride (bf16): 1200B = 16*75, odd q -> conflict-free b128
#define HBST 536           // sHb row stride (bf16): 1072B = 16*67, odd q -> conflict-free b128

__device__ __forceinline__ float fast_tanh(float x) {
#if __has_builtin(__builtin_amdgcn_tanhf)
  return __builtin_amdgcn_tanhf(x);
#else
  return tanhf(x);
#endif
}

__global__ __launch_bounds__(256) void prep_kernel(const float* __restrict__ w1,
                                                   const float* __restrict__ w2,
                                                   bf16* __restrict__ w1b,
                                                   bf16* __restrict__ w2b) {
  int i = blockIdx.x * 256 + threadIdx.x;
  if (i < Hh * KT) w1b[i] = (bf16)w1[i];
  if (i < 32 * Hh) {                       // W2 padded to 32 rows (rows 17..31 zero)
    int n = i >> 9;
    w2b[i] = (n < Vv) ? (bf16)w2[i] : (bf16)0.f;
  }
}

__global__ __launch_bounds__(256) void entity_rnn_kernel(
    const float* __restrict__ inputs, const float* __restrict__ embt,
    const float* __restrict__ b1,     const float* __restrict__ b2,
    const bf16*  __restrict__ w1b,    const bf16*  __restrict__ w2b,
    float* __restrict__ o_logits, float* __restrict__ o_preds,
    float* __restrict__ o_probs,  float* __restrict__ o_pe) {
  __shared__ __align__(16) bf16  sA [16 * AST];   // x_t(bf16) || prev_emb(bf16)   19200B
  __shared__ __align__(16) bf16  sHb[16 * HBST];  // h bf16, batch-major           17152B
  __shared__ float sB1[Hh];
  __shared__ float sB2[32];
  __shared__ float sEmb[Vv * Ee];                 // PAD row zeroed
  __shared__ float sLG[16 * Vv];
  __shared__ float sLP[16 * Vv];
  __shared__ int   sPrd[16];

  const int tid  = threadIdx.x;
  const int lane = tid & 31;
  const int wv   = tid >> 5;     // 8 waves
  const int b0   = blockIdx.x * 16;

  // ---- one-time LDS init ----
  for (int j = tid; j < Hh; j += 256) sB1[j] = b1[j];
  if (tid < 32) sB2[tid] = (tid < Vv) ? b2[tid] : 0.f;
  for (int j = tid; j < Vv * Ee; j += 256) {
    int r = j / Ee;
    sEmb[j] = (r == PADi) ? 0.f : embt[j];
  }
  for (int j = tid; j < 16 * Ee; j += 256)            // prev_emb init = table[PAD] = 0
    sA[(j >> 6) * AST + Dd + (j & 63)] = (bf16)0.f;
  __syncthreads();

  const int m16   = lane & 15;   // WMMA row / column lane id
  const int kh    = lane >> 4;   // lane half selects K sub-range
  const int nbase = wv * 64;     // GEMM1: each wave owns 64 H-columns (4 N-tiles)

  const bf16* wbase  = w1b + (size_t)(nbase + m16) * KT + 16 * kh;   // GEMM1 B base
  const bf16* w2base = w2b + (size_t)(wv * 16 + m16) * Hh + 16 * kh; // GEMM2 B base (waves 0/1)

  // fragment load helpers (bf16 WMMA lane layouts)
  auto loadA = [&](const bf16* arr, int stride, int k0) -> v16bf {
    const bf16* ap = arr + m16 * stride + k0 + 16 * kh;
    v8bf al = *(const v8bf*)ap;
    v8bf ah = *(const v8bf*)(ap + 8);
    return __builtin_shufflevector(al, ah, 0,1,2,3,4,5,6,7,8,9,10,11,12,13,14,15);
  };
  auto loadB = [&](const bf16* bp) -> v16bf {
    v8bf bl = *(const v8bf*)bp;
    v8bf bh = *(const v8bf*)(bp + 8);
    return __builtin_shufflevector(bl, bh, 0,1,2,3,4,5,6,7,8,9,10,11,12,13,14,15);
  };

  for (int t = 0; t < Ll; ++t) {
    // ---- stage x_t tile (f32 -> bf16) into sA[:, 0:512] ----
    for (int j = tid; j < 16 * (Dd / 4); j += 256) {
      int m = j >> 7, c4 = j & 127;
      v4f x = *(const v4f*)(inputs + ((size_t)(b0 + m) * Ll + t) * Dd + c4 * 4);
      v4bf xb;
      xb[0] = (bf16)x[0]; xb[1] = (bf16)x[1]; xb[2] = (bf16)x[2]; xb[3] = (bf16)x[3];
      *(v4bf*)&sA[m * AST + c4 * 4] = xb;
    }
    __syncthreads();

    // ---- GEMM1: [16,576] x W1^T[576,512], bf16 WMMA, f32 accum ----
    // MUST stay rolled (unroll(disable)): any further unrolling lets LICM
    // hoist the t-invariant W1 fragment loads out of the t-loop, which
    // spills ~576 VGPRs to scratch (observed in rounds 1 and 4).
    // Software-pipelined: fragments for k0+32 load while k0's WMMAs issue;
    // the rotation moves dual-issue and co-execute with the XDL WMMA pipe.
    v8f acc[4] = {v8f{}, v8f{}, v8f{}, v8f{}};
    v16bf af  = loadA(sA, AST, 0);
    v16bf bf0 = loadB(wbase + 0 * 16 * KT);
    v16bf bf1 = loadB(wbase + 1 * 16 * KT);
    v16bf bf2 = loadB(wbase + 2 * 16 * KT);
    v16bf bf3 = loadB(wbase + 3 * 16 * KT);
#pragma clang loop unroll(disable)
    for (int k0 = 32; k0 < KT; k0 += 32) {
      v16bf an  = loadA(sA, AST, k0);
      v16bf bn0 = loadB(wbase + k0 + 0 * 16 * KT);
      v16bf bn1 = loadB(wbase + k0 + 1 * 16 * KT);
      v16bf bn2 = loadB(wbase + k0 + 2 * 16 * KT);
      v16bf bn3 = loadB(wbase + k0 + 3 * 16 * KT);
      __builtin_prefetch((const void*)(wbase + k0 + 32), 0, 1);  // speculative; ws-resident
      acc[0] = __builtin_amdgcn_wmma_f32_16x16x32_bf16(false, af, false, bf0, (short)0, acc[0], false, false);
      acc[1] = __builtin_amdgcn_wmma_f32_16x16x32_bf16(false, af, false, bf1, (short)0, acc[1], false, false);
      acc[2] = __builtin_amdgcn_wmma_f32_16x16x32_bf16(false, af, false, bf2, (short)0, acc[2], false, false);
      acc[3] = __builtin_amdgcn_wmma_f32_16x16x32_bf16(false, af, false, bf3, (short)0, acc[3], false, false);
      af = an; bf0 = bn0; bf1 = bn1; bf2 = bn2; bf3 = bn3;
    }
    acc[0] = __builtin_amdgcn_wmma_f32_16x16x32_bf16(false, af, false, bf0, (short)0, acc[0], false, false);
    acc[1] = __builtin_amdgcn_wmma_f32_16x16x32_bf16(false, af, false, bf1, (short)0, acc[1], false, false);
    acc[2] = __builtin_amdgcn_wmma_f32_16x16x32_bf16(false, af, false, bf2, (short)0, acc[2], false, false);
    acc[3] = __builtin_amdgcn_wmma_f32_16x16x32_bf16(false, af, false, bf3, (short)0, acc[3], false, false);

    // ---- bias + tanh, store h as bf16 batch-major (A layout for GEMM2) ----
#pragma unroll
    for (int i = 0; i < 4; ++i) {
      int n = nbase + i * 16 + m16;
      float bb = sB1[n];
#pragma unroll
      for (int r = 0; r < 8; ++r) {
        int m = kh * 8 + r;   // C layout: lanes 0-15 -> M=r, lanes 16-31 -> M=r+8
        sHb[m * HBST + n] = (bf16)fast_tanh(acc[i][r] + bb);
      }
    }
    __syncthreads();

    // ---- GEMM2: logits[16, 32pad] = h(bf16) x W2pad^T via WMMA, waves 0-1 ----
    if (wv < 2) {                 // wave-uniform: EXEC all-ones inside
      v8f acc2 = {};
#pragma clang loop unroll(disable)
      for (int k0 = 0; k0 < Hh; k0 += 32) {
        v16bf a2 = loadA(sHb, HBST, k0);
        v16bf b2f = loadB(w2base + k0);
        acc2 = __builtin_amdgcn_wmma_f32_16x16x32_bf16(false, a2, false, b2f, (short)0, acc2, false, false);
      }
      int v = wv * 16 + m16;      // logit column
      if (v < Vv) {
        float bb = sB2[v];
#pragma unroll
        for (int r = 0; r < 8; ++r) {
          int m = kh * 8 + r;
          sLG[m * Vv + v] = acc2[r] + bb;
        }
      }
    }
    __syncthreads();

    // ---- log_softmax + argmax (one thread per row; V=17 is tiny) ----
    if (tid < 16) {
      const float* lg = &sLG[tid * Vv];
      float mx = lg[0]; int am = 0;
#pragma unroll
      for (int v = 1; v < Vv; ++v) { float l = lg[v]; if (l > mx) { mx = l; am = v; } }
      float s = 0.f;
#pragma unroll
      for (int v = 0; v < Vv; ++v) s += __expf(lg[v] - mx);
      float lse = mx + __logf(s);
#pragma unroll
      for (int v = 0; v < Vv; ++v) sLP[tid * Vv + v] = lg[v] - lse;
      sPrd[tid] = am;
    }
    __syncthreads();

    // ---- write outputs, feed back embedding for next step ----
    for (int j = tid; j < 16 * Vv; j += 256) {
      int m = j / Vv, v = j % Vv;
      size_t o = ((size_t)(b0 + m) * Ll + t) * Vv + v;
      o_logits[o] = sLG[m * Vv + v];
      o_probs[o]  = sLP[m * Vv + v];
    }
    if (tid < 16) o_preds[(size_t)(b0 + tid) * Ll + t] = (float)sPrd[tid];
    for (int j = tid; j < 16 * Ee; j += 256) {
      int m = j >> 6, e = j & 63;
      float ev = sEmb[sPrd[m] * Ee + e];            // PAD row already zeroed
      o_pe[((size_t)(b0 + m) * Ll + t) * Ee + e] = ev;
      sA[m * AST + Dd + e] = (bf16)ev;              // prev_emb for t+1
    }
    __syncthreads();
  }
}

extern "C" void kernel_launch(void* const* d_in, const int* in_sizes, int n_in,
                              void* d_out, int out_size, void* d_ws, size_t ws_size,
                              hipStream_t stream) {
  const float* inputs = (const float*)d_in[0];   // [B,L,D]
  const float* embt   = (const float*)d_in[1];   // [V,E]
  const float* W1     = (const float*)d_in[2];   // [H,D+E]
  const float* b1     = (const float*)d_in[3];   // [H]
  const float* W2     = (const float*)d_in[4];   // [V,H]
  const float* b2     = (const float*)d_in[5];   // [V]

  bf16* w1b = (bf16*)d_ws;                       // 512*576*2 = 576 KB
  bf16* w2b = w1b + (size_t)Hh * KT;             // 32*512*2  =  32 KB (padded W2)

  float* o_logits = (float*)d_out;                         // B*L*V
  float* o_preds  = o_logits + (size_t)Bb * Ll * Vv;       // B*L
  float* o_probs  = o_preds  + (size_t)Bb * Ll;            // B*L*V
  float* o_pe     = o_probs  + (size_t)Bb * Ll * Vv;       // B*L*E

  hipLaunchKernelGGL(prep_kernel, dim3((Hh * KT + 255) / 256), dim3(256), 0, stream,
                     W1, W2, w1b, w2b);
  hipLaunchKernelGGL(entity_rnn_kernel, dim3(Bb / 16), dim3(256), 0, stream,
                     inputs, embt, b1, b2, w1b, w2b,
                     o_logits, o_preds, o_probs, o_pe);
}
